// STGraphTCN_12987981103586
// MI455X (gfx1250) — compile-verified, matrix operand order checked by reading, not run on hardware
//
#include <hip/hip_runtime.h>
#include <hip/hip_bf16.h>
#include <math.h>

typedef _Float16 half_t;
typedef __attribute__((ext_vector_type(16))) _Float16 v16h;
typedef __attribute__((ext_vector_type(8)))  _Float16 v8h;
typedef __attribute__((ext_vector_type(8)))  float    v8f;

#define NBT  512   // B*T
#define NN   128   // nodes
#define SS   12
#define CC   32
#define DD   256
#define ETA_C  1.0f
#define BETA_C 0.6f
#define NWAVE 8

#define WMMA(a, b, c) __builtin_amdgcn_wmma_f32_16x16x32_f16(false, (a), false, (b), (short)0, (c), false, false)

// ---- wave reductions (wave32) ----
__device__ __forceinline__ float wred_sum(float v) {
#pragma unroll
  for (int o = 16; o > 0; o >>= 1) v += __shfl_xor(v, o, 32);
  return v;
}
__device__ __forceinline__ float wred_max(float v) {
#pragma unroll
  for (int o = 16; o > 0; o >>= 1) v = fmaxf(v, __shfl_xor(v, o, 32));
  return v;
}

// ---- WMMA fragment helpers (ISA 7.12.2 wave32 layouts) ----
// A-matrix 16x32 f16 from row-major LDS (ld in elements): two contiguous
// 8-half runs per lane -> 2x ds_load_b128.
__device__ __forceinline__ v16h frag_a_f16(const half_t* p, int ld) {
  const int lane = threadIdx.x & 31;
  const half_t* row = p + (lane & 15) * ld + ((lane < 16) ? 0 : 8);
  v8h lo = *(const v8h*)row;
  v8h hi = *(const v8h*)(row + 16);
  return __builtin_shufflevector(lo, hi, 0, 1, 2, 3, 4, 5, 6, 7, 8, 9, 10, 11, 12, 13, 14, 15);
}

// B-matrix fragment from a pre-swizzled slab: slab layout [nt][lane][16 halves],
// so each lane's fragment is 16 contiguous halves -> 2x ds_load_b128.
__device__ __forceinline__ v16h frag_slab(const half_t* slab, int nt) {
  const int lane = threadIdx.x & 31;
  const half_t* p = slab + (nt * 32 + lane) * 16;
  v8h lo = *(const v8h*)p;
  v8h hi = *(const v8h*)(p + 8);
  return __builtin_shufflevector(lo, hi, 0, 1, 2, 3, 4, 5, 6, 7, 8, 9, 10, 11, 12, 13, 14, 15);
}

// C/D 16x16 f32: lane<16 -> N=lane, M=i ; lane>=16 -> N=lane-16, M=i+8
__device__ __forceinline__ void store_c_f16(half_t* p, int ld, v8f c) {
  const int lane = threadIdx.x & 31;
  const int n    = lane & 15;
  const int mb   = (lane < 16) ? 0 : 8;
#pragma unroll
  for (int i = 0; i < 8; ++i) p[(mb + i) * ld + n] = (half_t)c[i];
}

// Swizzle map: element (k,n) of a 32x256 slab -> slab[(nt*32 + lane)*16 + j]
//   lane = (n&15) + ((k&8)?16:0) ; j = (k&7) + ((k&16)?8:0) ; nt = n>>4
// Stage from global f32 (coalesced float4 reads, 2048 float4 / 256 threads).
__device__ __forceinline__ void stage_slab_f32(const float* __restrict__ W,
                                               half_t* __restrict__ slab, int tid) {
  for (int e = tid; e < 2048; e += 256) {
    const float4 v = ((const float4*)W)[e];
    const int k   = e >> 6;
    const int n0  = (e & 63) * 4;
    const int j   = (k & 7) + ((k & 16) ? 8 : 0);
    const int lhi = (k & 8) ? 16 : 0;
    const int base = (((n0 >> 4) * 32) + lhi + (n0 & 15)) * 16 + j;
    slab[base]      = (half_t)v.x;
    slab[base + 16] = (half_t)v.y;
    slab[base + 32] = (half_t)v.z;
    slab[base + 48] = (half_t)v.w;
  }
}

// Stage from row-major f16 LDS (vector b128 reads, scatter b16 writes).
__device__ __forceinline__ void stage_slab_f16(const half_t* __restrict__ S,
                                               half_t* __restrict__ slab, int tid) {
  for (int e = tid; e < 1024; e += 256) {
    const int k  = e >> 5;
    const int n0 = (e & 31) * 8;
    v8h v = *(const v8h*)(S + k * 256 + n0);
    const int j   = (k & 7) + ((k & 16) ? 8 : 0);
    const int lhi = (k & 8) ? 16 : 0;
    int base = (((n0 >> 4) * 32) + lhi + (n0 & 15)) * 16 + j;
#pragma unroll
    for (int i = 0; i < 8; ++i) slab[base + i * 16] = v[i];
  }
}

__global__ __launch_bounds__(256)
void stgraph_fused(const float* __restrict__ X, const float* __restrict__ Msk,
                   const float* __restrict__ A_stat, const float* __restrict__ M_mask,
                   const float* __restrict__ reliab,
                   const float* __restrict__ Wproj, const float* __restrict__ Wq,
                   const float* __restrict__ Wk,    const float* __restrict__ Wprop,
                   const float* __restrict__ g1, const float* __restrict__ b1,
                   const float* __restrict__ g2, const float* __restrict__ b2,
                   float* __restrict__ outH, float* __restrict__ outA)
{
  extern __shared__ char smem[];
  half_t* HN16  = (half_t*)smem;                 // [128][256] f16  64 KB (Hn; slab region in P6)
  half_t* R2_16 = (half_t*)(smem + 65536);       // [128][256] f16  64 KB (K, then H1)
  char*   R3    = smem + 131072;                 //                 64 KB
  half_t* X16   = (half_t*)R3;                   // [128][32]  f16  8 KB  (P0/P1)
  half_t* WSLAB = (half_t*)(smem + 147456);      // [16][32][16] f16 16 KB (P1/P2 weight slab)
  float*  LOG   = (float*)R3;                    // [128][128] f32  (logits -> A_fuse f16 overlay)
  half_t* Q16   = (half_t*)(smem + 196608);      // [128][256] f16  64 KB (Q; slab region in P5)
  float*  HBUF  = (float*)(smem + 131072);       // [128][256] f32 128 KB (overlays LOG+Q16)

  const int bt   = blockIdx.x;
  const int tid  = threadIdx.x;
  const int lane = tid & 31;
  const int wave = tid >> 5;

  // warm L2 with the weight matrices (global_prefetch_b8)
  __builtin_prefetch(Wq    + tid * 256, 0, 0);
  __builtin_prefetch(Wk    + tid * 256, 0, 0);
  __builtin_prefetch(Wprop + tid * 256, 0, 0);

  // ---- P0: masked short-window mean over S; lane == channel (C==32==wave32) ----
  const float* Xb = X   + (size_t)bt * NN * SS * CC;
  const float* Mb = Msk + (size_t)bt * NN * SS;
  for (int n = wave; n < NN; n += NWAVE) {
    float ms = 0.f;
#pragma unroll
    for (int s = 0; s < SS; ++s) ms += Mb[n * SS + s];
    const float inv = 1.0f / fmaxf(ms, 1.0f);
    float acc = 0.f;
#pragma unroll
    for (int s = 0; s < SS; ++s) acc += Mb[n * SS + s] * Xb[(n * SS + s) * CC + lane];
    X16[n * CC + lane] = (half_t)(acc * inv);
  }
  // stage W_proj slab (K=32: one slab) while P0 finishes
  stage_slab_f32(Wproj, WSLAB, tid);
  __syncthreads();

  // ---- P1: Hn = x @ W_proj   [128,32]@[32,256], one WMMA per tile ----
  for (int t = wave; t < 8 * 16; t += NWAVE) {
    const int mt = t >> 4, nt = t & 15;
    v16h a = frag_a_f16(X16 + mt * 16 * CC, CC);
    v16h b = frag_slab(WSLAB, nt);
    v8f c = {};
    c = WMMA(a, b, c);
    store_c_f16(HN16 + (mt * 16) * DD + nt * 16, DD, c);
  }

  // ---- P2: Q = Hn@Wq -> Q16 ; K = Hn@Wk -> R2_16 (wave = M-tile, regs carry C) ----
  for (int pass = 0; pass < 2; ++pass) {
    const float* W  = pass ? Wk    : Wq;
    half_t*     dst = pass ? R2_16 : Q16;
    v8f acc[16];
#pragma unroll
    for (int nt = 0; nt < 16; ++nt) acc[nt] = (v8f){};
    for (int kb = 0; kb < 8; ++kb) {
      __syncthreads();                       // previous slab consumers done
      stage_slab_f32(W + kb * 32 * DD, WSLAB, tid);
      __syncthreads();
      v16h a = frag_a_f16(HN16 + (wave * 16) * DD + kb * 32, DD);
#pragma unroll
      for (int nt = 0; nt < 16; ++nt) {
        v16h b = frag_slab(WSLAB, nt);
        acc[nt] = WMMA(a, b, acc[nt]);
      }
    }
#pragma unroll
    for (int nt = 0; nt < 16; ++nt)
      store_c_f16(dst + (wave * 16) * DD + nt * 16, DD, acc[nt]);
  }
  __syncthreads();

  // ---- P3: logits = Q @ K^T / 16  (B^T pattern == A-layout on K rows) ----
  for (int t = wave; t < 64; t += NWAVE) {
    const int mt = t >> 3, nt = t & 7;
    v8f c = {};
#pragma unroll
    for (int kb = 0; kb < 8; ++kb) {
      v16h a  = frag_a_f16(Q16   + (mt * 16) * DD + kb * 32, DD);
      v16h bT = frag_a_f16(R2_16 + (nt * 16) * DD + kb * 32, DD);
      c = WMMA(a, bT, c);
    }
    const int n = lane & 15, mb = (lane < 16) ? 0 : 8;
    float* dst = LOG + (mt * 16) * NN + nt * 16;
#pragma unroll
    for (int i = 0; i < 8; ++i) dst[(mb + i) * NN + n] = c[i] * 0.0625f;  // 1/sqrt(256)
  }
  __syncthreads();

  // ---- P4: bias + mask + softmax + fuse + reliability + renorm (wave per row) ----
  {
    const float* rl = reliab + (size_t)bt * NN;
    for (int r = wave; r < NN; r += NWAVE) {
      const float* lrow = LOG + r * NN;
      const float* As = A_stat + r * NN;
      const float* Mm = M_mask + r * NN;
      const float rr = fminf(fmaxf(rl[r], 0.f), 1.f);
      float lv[4];
      float mx = -3.0e38f;
#pragma unroll
      for (int q = 0; q < 4; ++q) {
        const int j = lane + 32 * q;
        const float ac = fminf(fmaxf(As[j], 1e-3f), 1.f - 1e-3f);
        float l = lrow[j] + ETA_C * __logf(ac / (1.f - ac));
        if (Mm[j] == 0.f) l = -3.0e38f;
        lv[q] = l;
        mx = fmaxf(mx, l);
      }
      mx = wred_max(mx);
      float s = 0.f;
#pragma unroll
      for (int q = 0; q < 4; ++q) { lv[q] = __expf(lv[q] - mx); s += lv[q]; }
      s = wred_sum(s);
      const float inv = 1.f / s;
      float fs = 0.f;
#pragma unroll
      for (int q = 0; q < 4; ++q) {
        const int j = lane + 32 * q;
        const float rj = fminf(fmaxf(rl[j], 0.f), 1.f);
        lv[q] = (BETA_C * lv[q] * inv + (1.f - BETA_C) * As[j]) * rr * rj * Mm[j];
        fs += lv[q];
      }
      fs = wred_sum(fs);
      const float rn = 1.f / (fs + 1e-8f);
      half_t* arow = (half_t*)lrow;   // f16 overlay on the 512B row slot
      float*  oA   = outA + ((size_t)bt * NN + r) * NN;
#pragma unroll
      for (int q = 0; q < 4; ++q) {
        const int j = lane + 32 * q;
        const float v = lv[q] * rn;
        oA[j]   = v;
        arow[j] = (half_t)v;
      }
    }
  }

  // ---- P5: H1 = A_fuse @ Hn  [128,128]@[128,256] -> R2_16 (slab = dead Q16) ----
  {
    half_t* A16   = (half_t*)R3;   // row stride 256 halves
    half_t* SLAB5 = Q16;           // Q dead after P3
    v8f acc[16];
#pragma unroll
    for (int nt = 0; nt < 16; ++nt) acc[nt] = (v8f){};
    for (int kb = 0; kb < 4; ++kb) {
      __syncthreads();             // P4 writes / prev slab reads done
      stage_slab_f16(HN16 + (kb * 32) * DD, SLAB5, tid);
      __syncthreads();
      v16h a = frag_a_f16(A16 + (wave * 16) * 256 + kb * 32, 256);
#pragma unroll
      for (int nt = 0; nt < 16; ++nt) {
        v16h b = frag_slab(SLAB5, nt);
        acc[nt] = WMMA(a, b, acc[nt]);
      }
    }
#pragma unroll
    for (int nt = 0; nt < 16; ++nt)
      store_c_f16(R2_16 + (wave * 16) * DD + nt * 16, DD, acc[nt]);
  }

  // ---- P6: H2 = relu(H1 @ W_prop) -> HBUF f32 (slab = dead HN16 region) ----
  {
    half_t* SLAB6 = HN16;          // Hn dead after P5 staging
    v8f acc[16];
#pragma unroll
    for (int nt = 0; nt < 16; ++nt) acc[nt] = (v8f){};
    for (int kb = 0; kb < 8; ++kb) {
      __syncthreads();
      stage_slab_f32(Wprop + kb * 32 * DD, SLAB6, tid);
      __syncthreads();
      v16h a = frag_a_f16(R2_16 + (wave * 16) * DD + kb * 32, DD);
#pragma unroll
      for (int nt = 0; nt < 16; ++nt) {
        v16h b = frag_slab(SLAB6, nt);
        acc[nt] = WMMA(a, b, acc[nt]);
      }
    }
    __syncthreads();               // LOG/A16 + Q16 fully dead before HBUF writes
    const int n = lane & 15, mb = (lane < 16) ? 0 : 8;
#pragma unroll
    for (int nt = 0; nt < 16; ++nt) {
      float* dst = HBUF + (wave * 16) * DD + nt * 16;
#pragma unroll
      for (int i = 0; i < 8; ++i) dst[(mb + i) * DD + n] = fmaxf(acc[nt][i], 0.f);
    }
  }
  __syncthreads();

  // ---- P7: double LayerNorm over D=256 (wave per row, register resident) ----
  for (int r = wave; r < NN; r += NWAVE) {
    const float* h = HBUF + r * DD;
    float x[8];
    float mu = 0.f;
#pragma unroll
    for (int q = 0; q < 8; ++q) { x[q] = h[lane + 32 * q]; mu += x[q]; }
    mu = wred_sum(mu) * (1.f / DD);
    float var = 0.f;
#pragma unroll
    for (int q = 0; q < 8; ++q) { const float dv = x[q] - mu; var += dv * dv; }
    var = wred_sum(var) * (1.f / DD);
    const float is1 = rsqrtf(var + 1e-5f);
    float mu2 = 0.f;
#pragma unroll
    for (int q = 0; q < 8; ++q) {
      const int d = lane + 32 * q;
      x[q] = (x[q] - mu) * is1 * g1[d] + b1[d];
      mu2 += x[q];
    }
    mu2 = wred_sum(mu2) * (1.f / DD);
    float v2 = 0.f;
#pragma unroll
    for (int q = 0; q < 8; ++q) { const float dv = x[q] - mu2; v2 += dv * dv; }
    v2 = wred_sum(v2) * (1.f / DD);
    const float is2 = rsqrtf(v2 + 1e-5f);
    float* oH = outH + ((size_t)bt * NN + r) * DD;
#pragma unroll
    for (int q = 0; q < 8; ++q) {
      const int d = lane + 32 * q;
      oH[d] = (x[q] - mu2) * is2 * g2[d] + b2[d];
    }
  }
}

extern "C" void kernel_launch(void* const* d_in, const int* in_sizes, int n_in,
                              void* d_out, int out_size, void* d_ws, size_t ws_size,
                              hipStream_t stream) {
  (void)in_sizes; (void)n_in; (void)out_size; (void)d_ws; (void)ws_size;
  const float* X      = (const float*)d_in[0];
  const float* Msk    = (const float*)d_in[1];
  const float* A_stat = (const float*)d_in[2];
  const float* M_mask = (const float*)d_in[3];
  const float* reliab = (const float*)d_in[4];
  const float* Wproj  = (const float*)d_in[5];
  const float* Wq     = (const float*)d_in[6];
  const float* Wk     = (const float*)d_in[7];
  const float* Wprop  = (const float*)d_in[8];
  const float* g1     = (const float*)d_in[9];
  const float* b1     = (const float*)d_in[10];
  const float* g2     = (const float*)d_in[11];
  const float* b2     = (const float*)d_in[12];

  float* outH = (float*)d_out;
  float* outA = outH + (size_t)NBT * NN * DD;

  const size_t lds_bytes = 262144;  // 256 KB dynamic LDS (<= 320 KB/WGP)
  (void)hipFuncSetAttribute((const void*)stgraph_fused,
                            hipFuncAttributeMaxDynamicSharedMemorySize,
                            (int)lds_bytes);
  stgraph_fused<<<NBT, 256, lds_bytes, stream>>>(
      X, Msk, A_stat, M_mask, reliab, Wproj, Wq, Wk, Wprop,
      g1, b1, g2, b2, outH, outA);
}
